// feature_extractor_42021960024152
// MI455X (gfx1250) — matrix operand order repaired and checked by reading, not compiled, not run on hardware
//
#include <hip/hip_runtime.h>
#include <math.h>

// ---------------- problem constants ----------------
#define SRATE   22050
#define FL      1024
#define FP      256
#define XLEN    661500
#define NFRM    2584                 // frames per batch row
#define BATCH   8
#define MTOT    (BATCH * NFRM)       // 20672 = 323 * 64
#define NB      513                  // rfft bins
#define NPAD    544                  // bins padded to 17*32
#define DSB     232                  // truncated bins
#define LAGS_US 32                   // lags for order-30 levinson (need 31)
#define LAGS_DS 32                   // lags for order-10 levinson + tilt
#define LAGS_R  448                  // lags for pitch search (need 442)
#define LAG_MIN 44
#define LAG_MAX 441
#define TWO_PI  6.28318530717958647692f

typedef __attribute__((ext_vector_type(16))) _Float16 v16h;
typedef __attribute__((ext_vector_type(8)))  _Float16 v8h;
typedef __attribute__((ext_vector_type(8)))  float    v8f;

// B-matrix fragment swizzle (contiguous k order matching the A fragment, not the ISA table order; 16-bit B 32x16 per wave):
// element (k,n) of a KxN matrix lives at tile (kt = k/32, nt = n/16),
// lane = (k%32/16)*16 + (n%16), half h = k%16; 16 halves per lane.
__device__ inline size_t bswiz(int k, int n, int N) {
  int kt = k >> 5, ki = k & 31;
  int lane = (ki >> 4) * 16 + (n & 15);
  int h = ki & 15;
  return (((size_t)kt * (N >> 4) + (n >> 4)) * 32 + lane) * 16 + h;
}

// ---------------- window build (Blackman, energy-normalized) ----------------
__global__ void win_kernel(float* __restrict__ win) {
  __shared__ float s[FL];
  int n = threadIdx.x;
  float ph = (float)n / (float)(FL - 1);
  float w = 0.42f - 0.5f * cosf(TWO_PI * ph) + 0.08f * cosf(2.0f * TWO_PI * ph);
  s[n] = w * w;
  __syncthreads();
  for (int off = FL / 2; off > 0; off >>= 1) {
    if (n < off) s[n] += s[n + off];
    __syncthreads();
  }
  win[n] = w * rsqrtf(s[0]);
}

// ---------------- framing: raw + windowed frames in f16 ----------------
__global__ void frame_kernel(const float* __restrict__ x, const float* __restrict__ win,
                             _Float16* __restrict__ fw, _Float16* __restrict__ fr) {
  size_t idx = (size_t)blockIdx.x * blockDim.x + threadIdx.x;  // over MTOT*FL
  int m = (int)(idx >> 10);
  int n = (int)(idx & (FL - 1));
  int b = m / NFRM, f = m % NFRM;
  long src = (long)f * FP + n - FL / 2;
  float v = (src >= 0 && src < XLEN) ? x[(long)b * XLEN + src] : 0.0f;
  fr[idx] = (_Float16)v;
  fw[idx] = (_Float16)(v * win[n]);
}

// ---------------- forward DFT basis, written pre-swizzled ----------------
__global__ void fbasis_kernel(_Float16* __restrict__ cosF, _Float16* __restrict__ sinF) {
  size_t idx = (size_t)blockIdx.x * blockDim.x + threadIdx.x;  // FL*NPAD
  int k = (int)(idx / NPAD), n = (int)(idx % NPAD);
  float c = 0.0f, s = 0.0f;
  if (n < NB) {
    int kn = (k * n) & (FL - 1);                 // phase mod 1024
    float ang = TWO_PI * (float)kn / (float)FL;
    c = cosf(ang);
    s = -sinf(ang);
  }
  size_t o = bswiz(k, n, NPAD);
  cosF[o] = (_Float16)c;
  sinF[o] = (_Float16)s;
}

// ---------------- inverse-DFT (autocorrelation) basis, pre-swizzled ----------------
__global__ void ibasis_kernel(_Float16* __restrict__ out, int nbValid, int period, int lags) {
  size_t idx = (size_t)blockIdx.x * blockDim.x + threadIdx.x;  // NPAD*lags
  int bin = (int)(idx / lags), l = (int)(idx % lags);
  float v = 0.0f;
  if (bin < nbValid) {
    float wgt = (bin == 0 || bin == nbValid - 1) ? 1.0f : 2.0f;
    float ang = TWO_PI * fmodf((float)bin * (float)l, (float)period) / (float)period;
    v = wgt * cosf(ang) / (float)period;
  }
  out[bswiz(bin, l, lags)] = (_Float16)v;
}

// ---------------- WMMA f16 GEMM: C[MTOT x N] = A[MTOT x K] * Bsw[K x N swizzled] ----
// grid = (N/32, MTOT/64), block = 128 (4 waves). Wave w: rows [m0+16w, +16),
// cols [n0, n0+32) via two 16-wide B fragments (2 WMMAs / k-tile).
// A tile (64x32 f16) staged to LDS with global_load_async_to_lds_b128 (ASYNCcnt).
__global__ __launch_bounds__(128)
void gemm_wmma_kernel(const _Float16* __restrict__ A, const _Float16* __restrict__ Bsw,
                      float* __restrict__ C, int K, int N) {
  __shared__ _Float16 As[64 * 40];               // 64 rows, stride 40 halves (80B)
  const int tid  = threadIdx.x;
  const int lane = tid & 31;
  const int wv   = tid >> 5;
  const int m0 = blockIdx.y * 64, n0 = blockIdx.x * 32;
  const int ntile16 = n0 >> 4;                   // base 16-col fragment index

  // async-copy addressing: thread moves 2x16B of one row
  const int arow = tid >> 1;                     // 0..63
  const int acol = (tid & 1) * 16;               // 0 or 16 (halves)
  const unsigned ldsoff = (unsigned)(arow * 40 + acol) * 2u;  // As assumed at LDS off 0

  // A-fragment addressing (contiguous k = 16*(lane/16) + s, same order as B)
  const int frow = wv * 16 + (lane & 15);
  const int fkb  = (lane >> 4) * 16;

  v8f acc0 = {0.f, 0.f, 0.f, 0.f, 0.f, 0.f, 0.f, 0.f};
  v8f acc1 = {0.f, 0.f, 0.f, 0.f, 0.f, 0.f, 0.f, 0.f};

  const int nbfrag = N >> 4;
  for (int k0 = 0; k0 < K; k0 += 32) {
    __syncthreads();                             // previous tile fully consumed
    const _Float16* gsrc = A + (size_t)(m0 + arow) * K + k0 + acol;
    asm volatile("global_load_async_to_lds_b128 %0, %1, off"
                 :: "v"(ldsoff), "v"(gsrc) : "memory");
    asm volatile("global_load_async_to_lds_b128 %0, %1, off"
                 :: "v"(ldsoff + 16u), "v"(gsrc + 8) : "memory");
    asm volatile("s_wait_asynccnt 0x0" ::: "memory");
    __syncthreads();                             // tile visible to all waves

    v8h alo = *(const v8h*)(&As[frow * 40 + fkb]);
    v8h ahi = *(const v8h*)(&As[frow * 40 + fkb + 8]);
    v16h afrag = __builtin_shufflevector(alo, ahi, 0, 1, 2, 3, 4, 5, 6, 7,
                                         8, 9, 10, 11, 12, 13, 14, 15);

    const int kt = k0 >> 5;
    const v16h* bp = (const v16h*)Bsw + ((size_t)kt * nbfrag + ntile16) * 32 + lane;
    v16h b0 = bp[0];                             // cols n0   .. n0+16
    v16h b1 = bp[32];                            // cols n0+16.. n0+32

    acc0 = __builtin_amdgcn_wmma_f32_16x16x32_f16(false, afrag, false, b0,
                                                  (short)0, acc0, false, false);
    acc1 = __builtin_amdgcn_wmma_f32_16x16x32_f16(false, afrag, false, b1,
                                                  (short)0, acc1, false, false);
  }

  const int crow  = m0 + wv * 16 + (lane >> 4) * 8;
  const int ccol0 = n0 + (lane & 15);
#pragma unroll
  for (int r = 0; r < 8; ++r) {
    C[(size_t)(crow + r) * N + ccol0]      = acc0[r];
    C[(size_t)(crow + r) * N + ccol0 + 16] = acc1[r];
  }
}

// ---------------- power spectrum (+ optional x_ds complex output) ----------------
__global__ void power_kernel(const float* __restrict__ re, const float* __restrict__ im,
                             _Float16* __restrict__ p16, float* __restrict__ xds) {
  size_t idx = (size_t)blockIdx.x * blockDim.x + threadIdx.x;  // MTOT*NPAD
  int m = (int)(idx / NPAD), bin = (int)(idx % NPAD);
  float r = re[idx], i = im[idx];
  float p = (bin < NB) ? (r * r + i * i) : 0.0f;
  p16[idx] = (_Float16)p;
  if (xds != nullptr && bin < DSB) {
    size_t o = ((size_t)m * DSB + bin) * 2;
    xds[o] = r;
    xds[o + 1] = i;
  }
}

// ---------------- Levinson-Durbin (fully unrolled, registers only) ----------------
template <int ORDER>
__device__ inline void levinson(const float* __restrict__ r, float* __restrict__ A,
                                float* __restrict__ Kc) {
  float P[ORDER + 1];
#pragma unroll
  for (int i = 0; i <= ORDER; i++) A[i] = 0.0f;
  A[0] = 1.0f;
  float E = r[0];
#pragma unroll
  for (int m = 1; m <= ORDER; m++) {
    float acc = r[m];
#pragma unroll
    for (int j = 1; j < m; j++) acc += A[j] * r[m - j];
    float k = -acc / (E + 1e-9f);
#pragma unroll
    for (int i = 0; i <= ORDER; i++) P[i] = A[i];
    A[m] = k;
#pragma unroll
    for (int j = 1; j < m; j++) A[j] += k * P[m - j];
    E *= (1.0f - k * k);
    Kc[m - 1] = k;
  }
}

__global__ void levinson_kernel(const float* __restrict__ acD, const float* __restrict__ acW,
                                float* __restrict__ apenv, float* __restrict__ tilt_out,
                                float* __restrict__ rcoef_out) {
  int m = blockIdx.x * blockDim.x + threadIdx.x;
  if (m >= MTOT) return;
  float rD[11], A10[11], K10[10];
#pragma unroll
  for (int i = 0; i < 11; i++) rD[i] = acD[(size_t)m * LAGS_DS + i];
  levinson<10>(rD, A10, K10);
#pragma unroll
  for (int i = 0; i < 11; i++) apenv[(size_t)m * 11 + i] = A10[i];
  tilt_out[m] = rD[1] / (rD[0] + 1e-10f);

  float rW[31], A30[31], K30[30];
#pragma unroll
  for (int i = 0; i < 31; i++) rW[i] = acW[(size_t)m * LAGS_US + i];
  levinson<30>(rW, A30, K30);
#pragma unroll
  for (int i = 0; i < 30; i++) rcoef_out[(size_t)m * 30 + i] = K30[i];
}

// ---------------- Durand-Kerner roots + formants ----------------
__global__ void dk_kernel(const float* __restrict__ apenv, float* __restrict__ formants) {
  int m = blockIdx.x * blockDim.x + threadIdx.x;
  if (m >= MTOT) return;
  float c[11];
#pragma unroll
  for (int i = 0; i < 11; i++) c[i] = apenv[(size_t)m * 11 + i];  // c[0] == 1

  float zr[10], zi[10];
  zr[0] = 0.4f; zi[0] = 0.9f;
#pragma unroll
  for (int j = 1; j < 10; j++) {
    zr[j] = zr[j - 1] * 0.4f - zi[j - 1] * 0.9f;
    zi[j] = zr[j - 1] * 0.9f + zi[j - 1] * 0.4f;
  }

  for (int it = 0; it < 60; ++it) {
    float ozr[10], ozi[10];
#pragma unroll
    for (int j = 0; j < 10; j++) { ozr[j] = zr[j]; ozi[j] = zi[j]; }
#pragma unroll
    for (int i = 0; i < 10; i++) {
      float pr = c[0], pi = 0.0f;
#pragma unroll
      for (int t = 1; t <= 10; t++) {
        float nr = pr * ozr[i] - pi * ozi[i] + c[t];
        float ni = pr * ozi[i] + pi * ozr[i];
        pr = nr; pi = ni;
      }
      float dr = 1.0f, di = 0.0f;
#pragma unroll
      for (int j = 0; j < 10; j++) {
        float er = ozr[i] - ozr[j] + (i == j ? 1.0f : 0.0f);
        float ei = ozi[i] - ozi[j];
        float nr = dr * er - di * ei;
        float ni = dr * ei + di * er;
        dr = nr; di = ni;
      }
      dr += 1e-12f;
      float den = dr * dr + di * di + 1e-20f;
      float qr = (pr * dr + pi * di) / den;
      float qi = (pi * dr - pr * di) / den;
      zr[i] = ozr[i] - qr;
      zi[i] = ozi[i] - qi;
    }
  }

  float fq[10];
#pragma unroll
  for (int i = 0; i < 10; i++) {
    float ang = atan2f(zi[i], zr[i]);
    float f = ang * (10000.0f / TWO_PI);
    fq[i] = (ang > 1e-4f) ? f : 5000.0f;
  }
#pragma unroll
  for (int p = 0; p < 9; p++) {
#pragma unroll
    for (int j = 0; j < 9 - p; j++) {
      float a = fq[j], b = fq[j + 1];
      fq[j] = fminf(a, b);
      fq[j + 1] = fmaxf(a, b);
    }
  }
#pragma unroll
  for (int i = 0; i < 4; i++) formants[(size_t)m * 4 + i] = fq[i];
}

// ---------------- wave-level reductions ----------------
__device__ inline float wave_sum(float v) {
#pragma unroll
  for (int o = 16; o > 0; o >>= 1) v += __shfl_xor(v, o, 32);
  return v;
}

__global__ void energy_kernel(const float* __restrict__ x, float* __restrict__ eout) {
  int m = blockIdx.x * (blockDim.x >> 5) + (threadIdx.x >> 5);
  int lane = threadIdx.x & 31;
  if (m >= MTOT) return;
  int b = m / NFRM, f = m % NFRM;
  float s = 0.0f;
  for (int n = lane; n < FL; n += 32) {
    long src = (long)f * FP + n - FL / 2;
    float v = (src >= 0 && src < XLEN) ? x[(long)b * XLEN + src] : 0.0f;
    s += v * v;
  }
  s = wave_sum(s);
  if (lane == 0) eout[m] = 10.0f * log10f(s + 1e-10f);
}

__global__ void centroid_kernel(const float* __restrict__ rr, const float* __restrict__ ri,
                                float* __restrict__ cout) {
  int m = blockIdx.x * (blockDim.x >> 5) + (threadIdx.x >> 5);
  int lane = threadIdx.x & 31;
  if (m >= MTOT) return;
  const float* pr = rr + (size_t)m * NPAD;
  const float* pi = ri + (size_t)m * NPAD;
  float num = 0.0f, den = 0.0f;
  const float fstep = (float)SRATE / (float)FL;
  for (int bin = lane; bin < NB; bin += 32) {
    float a = pr[bin], b = pi[bin];
    float mag = sqrtf(a * a + b * b);
    num += mag * (fstep * (float)bin);
    den += mag;
  }
  num = wave_sum(num);
  den = wave_sum(den);
  if (lane == 0) cout[m] = num / (den + 1e-10f);
}

__global__ void pitch_kernel(const float* __restrict__ acR, float* __restrict__ pout,
                             float* __restrict__ vout, float* __restrict__ nout) {
  int m = blockIdx.x * (blockDim.x >> 5) + (threadIdx.x >> 5);
  int lane = threadIdx.x & 31;
  if (m >= MTOT) return;
  const float* a = acR + (size_t)m * LAGS_R;
  float bv = -1e30f;
  int bi = LAG_MIN;
  for (int l = LAG_MIN + lane; l <= LAG_MAX; l += 32) {
    float v = a[l];
    if (v > bv || (v == bv && l < bi)) { bv = v; bi = l; }
  }
#pragma unroll
  for (int o = 16; o > 0; o >>= 1) {
    float ov = __shfl_xor(bv, o, 32);
    int oi = __shfl_xor(bi, o, 32);
    if (ov > bv || (ov == bv && oi < bi)) { bv = ov; bi = oi; }
  }
  if (lane == 0) {
    float nccf = bv / (a[0] + 1e-9f);
    pout[m] = logf((float)SRATE / (float)bi);
    vout[m] = (nccf > 0.3f) ? 1.0f : 0.0f;
    nout[m] = nccf;
  }
}

// ---------------- host orchestration ----------------
extern "C" void kernel_launch(void* const* d_in, const int* in_sizes, int n_in,
                              void* d_out, int out_size, void* d_ws, size_t ws_size,
                              hipStream_t stream) {
  (void)in_sizes; (void)n_in; (void)out_size; (void)ws_size;
  const float* x = (const float*)d_in[0];
  float* out = (float*)d_out;

  // workspace carve-up (256B aligned)
  char* base = (char*)d_ws;
  size_t off = 0;
  auto alloc = [&](size_t bytes) -> void* {
    void* p = base + off;
    off = (off + bytes + 255) & ~(size_t)255;
    return p;
  };
  _Float16* fw16  = (_Float16*)alloc((size_t)MTOT * FL * 2);
  _Float16* fr16  = (_Float16*)alloc((size_t)MTOT * FL * 2);
  float*    sWr   = (float*)alloc((size_t)MTOT * NPAD * 4);
  float*    sWi   = (float*)alloc((size_t)MTOT * NPAD * 4);
  float*    sRr   = (float*)alloc((size_t)MTOT * NPAD * 4);
  float*    sRi   = (float*)alloc((size_t)MTOT * NPAD * 4);
  _Float16* pW16  = (_Float16*)alloc((size_t)MTOT * NPAD * 2);
  _Float16* pR16  = (_Float16*)alloc((size_t)MTOT * NPAD * 2);
  float*    acW   = (float*)alloc((size_t)MTOT * LAGS_US * 4);
  float*    acD   = (float*)alloc((size_t)MTOT * LAGS_DS * 4);
  float*    acR   = (float*)alloc((size_t)MTOT * LAGS_R * 4);
  _Float16* cosF  = (_Float16*)alloc((size_t)FL * NPAD * 2);
  _Float16* sinF  = (_Float16*)alloc((size_t)FL * NPAD * 2);
  _Float16* idftW = (_Float16*)alloc((size_t)NPAD * LAGS_US * 2);
  _Float16* idftD = (_Float16*)alloc((size_t)NPAD * LAGS_DS * 2);
  _Float16* idftR = (_Float16*)alloc((size_t)NPAD * LAGS_R * 2);
  float*    win   = (float*)alloc(FL * 4);
  float*    apenv = (float*)alloc((size_t)MTOT * 11 * 4);

  // output slices (reference tuple order)
  float* out_form = out;                               // (M,4)
  float* out_en   = out_form + (size_t)4 * MTOT;       // (M,)
  float* out_cen  = out_en + MTOT;
  float* out_tilt = out_cen + MTOT;
  float* out_pit  = out_tilt + MTOT;
  float* out_voi  = out_pit + MTOT;
  float* out_rc   = out_voi + MTOT;                    // (M,30)
  float* out_xds  = out_rc + (size_t)30 * MTOT;        // (M,232) complex -> re,im
  float* out_nccf = out_xds + (size_t)2 * DSB * MTOT;  // (M,)

  // 1. window + frames + swizzled bases
  hipLaunchKernelGGL(win_kernel, dim3(1), dim3(FL), 0, stream, win);
  hipLaunchKernelGGL(frame_kernel, dim3((MTOT * FL) / 256), dim3(256), 0, stream,
                     x, win, fw16, fr16);
  hipLaunchKernelGGL(fbasis_kernel, dim3((FL * NPAD) / 256), dim3(256), 0, stream, cosF, sinF);
  hipLaunchKernelGGL(ibasis_kernel, dim3((NPAD * LAGS_US) / 256), dim3(256), 0, stream,
                     idftW, NB, FL, LAGS_US);
  hipLaunchKernelGGL(ibasis_kernel, dim3((NPAD * LAGS_DS) / 256), dim3(256), 0, stream,
                     idftD, DSB, 2 * (DSB - 1), LAGS_DS);
  hipLaunchKernelGGL(ibasis_kernel, dim3((NPAD * LAGS_R) / 256), dim3(256), 0, stream,
                     idftR, NB, FL, LAGS_R);

  // 2. forward DFTs as WMMA GEMMs: M=20672, K=1024, N=544
  dim3 gSpec(NPAD / 32, MTOT / 64);
  hipLaunchKernelGGL(gemm_wmma_kernel, gSpec, dim3(128), 0, stream, fw16, cosF, sWr, FL, NPAD);
  hipLaunchKernelGGL(gemm_wmma_kernel, gSpec, dim3(128), 0, stream, fw16, sinF, sWi, FL, NPAD);
  hipLaunchKernelGGL(gemm_wmma_kernel, gSpec, dim3(128), 0, stream, fr16, cosF, sRr, FL, NPAD);
  hipLaunchKernelGGL(gemm_wmma_kernel, gSpec, dim3(128), 0, stream, fr16, sinF, sRi, FL, NPAD);

  // 3. power spectra (windowed one also emits x_ds)
  hipLaunchKernelGGL(power_kernel, dim3((MTOT * NPAD) / 256), dim3(256), 0, stream,
                     sWr, sWi, pW16, out_xds);
  hipLaunchKernelGGL(power_kernel, dim3((MTOT * NPAD) / 256), dim3(256), 0, stream,
                     sRr, sRi, pR16, (float*)nullptr);

  // 4. autocorrelations as WMMA GEMMs: K=544
  hipLaunchKernelGGL(gemm_wmma_kernel, dim3(LAGS_US / 32, MTOT / 64), dim3(128), 0, stream,
                     pW16, idftW, acW, NPAD, LAGS_US);
  hipLaunchKernelGGL(gemm_wmma_kernel, dim3(LAGS_DS / 32, MTOT / 64), dim3(128), 0, stream,
                     pW16, idftD, acD, NPAD, LAGS_DS);
  hipLaunchKernelGGL(gemm_wmma_kernel, dim3(LAGS_R / 32, MTOT / 64), dim3(128), 0, stream,
                     pR16, idftR, acR, NPAD, LAGS_R);

  // 5. per-frame features
  hipLaunchKernelGGL(levinson_kernel, dim3((MTOT + 127) / 128), dim3(128), 0, stream,
                     acD, acW, apenv, out_tilt, out_rc);
  hipLaunchKernelGGL(dk_kernel, dim3((MTOT + 127) / 128), dim3(128), 0, stream,
                     apenv, out_form);
  hipLaunchKernelGGL(energy_kernel, dim3(MTOT / 8), dim3(256), 0, stream, x, out_en);
  hipLaunchKernelGGL(centroid_kernel, dim3(MTOT / 8), dim3(256), 0, stream, sRr, sRi, out_cen);
  hipLaunchKernelGGL(pitch_kernel, dim3(MTOT / 8), dim3(256), 0, stream,
                     acR, out_pit, out_voi, out_nccf);
}
